// Attention_369367188096
// MI455X (gfx1250) — compile-verified, matrix operand order checked by reading, not run on hardware
//
#include <hip/hip_runtime.h>
#include <hip/hip_bf16.h>

// ---------------------------------------------------------------------------
// Attention block for MI455X (gfx1250), bf16 WMMA pipeline.
//   B=2, N=4096, IN_DIM=1024, DIM=1024, HEADS=8, d=128, MAX_MASK=2048
//   SCALE = DIM^-0.5 = 0.03125
// Kernel 1: qkv = x @ W_qkv         (f32 -> bf16 WMMA -> bf16 q/k/v [b,h,n,d])
// Kernel 2: flash attention          (bf16 WMMA QK^T + PV, f32 online softmax)
// Kernel 3: out = o @ W_out + b_out  (bf16 WMMA -> f32)
// ---------------------------------------------------------------------------

#define BB 2
#define NN 4096
#define INDIM 1024
#define DDIM 1024
#define HEADS 8
#define HD 128
#define K3 3072
#define MROWS (BB * NN)        // 8192
#define MAXMASK 2048
#define SCALE_ 0.03125f
#define L2E_ 1.44269504088896f

typedef __attribute__((ext_vector_type(16))) __bf16 v16bf;
typedef __attribute__((ext_vector_type(8)))  float  v8f;

__device__ inline __bf16 f2bf(float f) {
  unsigned u = __builtin_bit_cast(unsigned, f);
  u += 0x7FFFu + ((u >> 16) & 1u);               // round-to-nearest-even
  unsigned short s = (unsigned short)(u >> 16);
  return __builtin_bit_cast(__bf16, s);
}

__device__ inline v8f wmma_bf16(v16bf a, v16bf b, v8f c) {
  // v_wmma_f32_16x16x32_bf16  D = A(16x32) * B(32x16) + C
  return __builtin_amdgcn_wmma_f32_16x16x32_bf16(
      /*neg_a=*/false, a, /*neg_b=*/false, b,
      /*c_mod=*/(short)0, c, /*reuse_a=*/false, /*reuse_b=*/false);
}

// A-fragment (16 rows x 32 K), row-major LDS tile with element stride `stride`.
// Lane L: row = L&15; K = (L>=16 ? 8:0)+e and +16 for upper half of the vector.
__device__ inline v16bf ldsA_frag(const __bf16* base, int stride, int lane) {
  const int m  = lane & 15;
  const int kb = (lane >> 4) << 3;
  const __bf16* p = base + m * stride + kb;
  v16bf a;
#pragma unroll
  for (int e = 0; e < 8; ++e) { a[e] = p[e]; a[e + 8] = p[16 + e]; }
  return a;
}

// B-fragment (32 K x 16 N), from row-major [K][stride] LDS tile.
// Lane L: col = L&15; K = (L>=16 ? 8:0)+e and +16 for upper half.
__device__ inline v16bf ldsB_frag(const __bf16* base, int stride, int lane) {
  const int n  = lane & 15;
  const int kb = (lane >> 4) << 3;
  const __bf16* p = base + kb * stride + n;
  v16bf b;
#pragma unroll
  for (int e = 0; e < 8; ++e) { b[e] = p[e * stride]; b[e + 8] = p[(16 + e) * stride]; }
  return b;
}

// ---------------------------------------------------------------------------
// Kernel 1: QKV projection. Grid (3072/64, 8192/128), block 256 (8 waves).
// Workgroup tile 128x64, wave tile 32x32 (2x2 WMMA), K step 32.
// ---------------------------------------------------------------------------
__global__ __launch_bounds__(256)
void qkv_kernel(const float* __restrict__ x, const float* __restrict__ wqkv,
                __bf16* __restrict__ qo, __bf16* __restrict__ ko,
                __bf16* __restrict__ vo) {
  const int tid  = threadIdx.x;
  const int lane = tid & 31;
  const int wave = tid >> 5;
  const int half = lane >> 4;
  const int ln   = lane & 15;
  const int wm   = wave & 3;   // 0..3 -> 32-row strips
  const int wn   = wave >> 2;  // 0..1 -> 32-col strips
  const int rowBase = blockIdx.y * 128;
  const int colBase = blockIdx.x * 64;

  __shared__ __bf16 smA[128 * 40];  // 128x32, stride 40
  __shared__ __bf16 smB[32 * 72];   // 32x64, stride 72

  v8f acc[2][2] = {};

  for (int k0 = 0; k0 < INDIM; k0 += 32) {
    __syncthreads();
    { // A tile: 128x32 f32 -> bf16 (16 elems / thread)
      const int row = tid >> 1;
      const int col = (tid & 1) << 4;
      const float4* src =
          reinterpret_cast<const float4*>(x + (size_t)(rowBase + row) * INDIM + k0 + col);
#pragma unroll
      for (int q4 = 0; q4 < 4; ++q4) {
        float4 f = src[q4];
        __bf16* d = &smA[row * 40 + col + q4 * 4];
        d[0] = f2bf(f.x); d[1] = f2bf(f.y); d[2] = f2bf(f.z); d[3] = f2bf(f.w);
      }
      if (k0 + 32 < INDIM)
        __builtin_prefetch(x + (size_t)(rowBase + row) * INDIM + k0 + 32 + col, 0, 1);
    }
    { // B tile: 32x64 f32 -> bf16 (8 elems / thread)
      const int row = tid >> 3;
      const int col = (tid & 7) << 3;
      const float4* src =
          reinterpret_cast<const float4*>(wqkv + (size_t)(k0 + row) * K3 + colBase + col);
#pragma unroll
      for (int q4 = 0; q4 < 2; ++q4) {
        float4 f = src[q4];
        __bf16* d = &smB[row * 72 + col + q4 * 4];
        d[0] = f2bf(f.x); d[1] = f2bf(f.y); d[2] = f2bf(f.z); d[3] = f2bf(f.w);
      }
      if (k0 + 32 < INDIM)
        __builtin_prefetch(wqkv + (size_t)(k0 + 32 + row) * K3 + colBase + col, 0, 1);
    }
    __syncthreads();

    v16bf af[2], bfr[2];
#pragma unroll
    for (int i = 0; i < 2; ++i) af[i] = ldsA_frag(&smA[(wm * 32 + i * 16) * 40], 40, lane);
#pragma unroll
    for (int j = 0; j < 2; ++j) bfr[j] = ldsB_frag(&smB[wn * 32 + j * 16], 72, lane);
#pragma unroll
    for (int i = 0; i < 2; ++i)
#pragma unroll
      for (int j = 0; j < 2; ++j) acc[i][j] = wmma_bf16(af[i], bfr[j], acc[i][j]);
  }

  // Epilogue: scatter into q/k/v [b,h,n,d] bf16 (fused reshape/transpose).
  const int hh8 = half << 3;
#pragma unroll
  for (int i = 0; i < 2; ++i)
#pragma unroll
    for (int j = 0; j < 2; ++j) {
      const int c     = colBase + wn * 32 + j * 16 + ln;
      const int which = c >> 10;        // 0=q 1=k 2=v
      const int rem   = c & 1023;
      const int h_    = rem >> 7;
      const int dd    = rem & 127;
      __bf16* dst = (which == 0) ? qo : ((which == 1) ? ko : vo);
#pragma unroll
      for (int r = 0; r < 8; ++r) {
        const int m  = rowBase + wm * 32 + i * 16 + r + hh8;
        const int b_ = m >> 12;
        const int n_ = m & 4095;
        dst[(((size_t)(b_ * HEADS + h_)) * NN + n_) * HD + dd] = f2bf(acc[i][j][r]);
      }
    }
}

// ---------------------------------------------------------------------------
// Kernel 2: flash attention. Grid (B*H, N/128), block 256 (8 waves x 16 rows).
// Key blocks of 32; S via 8 WMMA, PV via 8 WMMA; f32 online softmax.
// ---------------------------------------------------------------------------
__global__ __launch_bounds__(256)
void attn_kernel(const __bf16* __restrict__ q, const __bf16* __restrict__ k,
                 const __bf16* __restrict__ v, const int* __restrict__ mask,
                 __bf16* __restrict__ o) {
  const int bh   = blockIdx.x;       // 0..15  (b*8 + h)
  const int qb   = blockIdx.y * 128; // query row base
  const int tid  = threadIdx.x;
  const int lane = tid & 31;
  const int wave = tid >> 5;
  const int half = lane >> 4;
  const int ln   = lane & 15;

  const __bf16* qbase = q + (size_t)bh * NN * HD;
  const __bf16* kbase = k + (size_t)bh * NN * HD;
  const __bf16* vbase = v + (size_t)bh * NN * HD;

  __shared__ __bf16 smK[32 * 136];     // [key][d], stride 136
  __shared__ __bf16 smV[32 * 136];
  __shared__ __bf16 smP[8][16 * 40];   // per-wave P transpose scratch

  // Resident Q fragments: 16 rows x 128 d, 4 K-chunks of 32.
  const int qw = qb + wave * 16;
  v16bf qfrag[4];
  {
    const int kb = half << 3;
    const __bf16* qp = qbase + (size_t)(qw + ln) * HD;
#pragma unroll
    for (int c = 0; c < 4; ++c)
#pragma unroll
      for (int e = 0; e < 8; ++e) {
        qfrag[c][e]     = qp[c * 32 + kb + e];
        qfrag[c][e + 8] = qp[c * 32 + 16 + kb + e];
      }
  }

  v8f oacc[8] = {};
  float row_m[8], row_l[8];
#pragma unroll
  for (int r = 0; r < 8; ++r) { row_m[r] = -3.0e38f; row_l[r] = 0.0f; }

  for (int jb = 0; jb < NN; jb += 32) {
    __syncthreads();
    { // cooperative K/V tile loads: 32 keys x 128 d, 16 bf16 per thread
      const int row = tid >> 3;
      const int col = (tid & 7) << 4;
      const int4* ks = reinterpret_cast<const int4*>(kbase + (size_t)(jb + row) * HD + col);
      const int4* vs = reinterpret_cast<const int4*>(vbase + (size_t)(jb + row) * HD + col);
      int4 k0 = ks[0], k1 = ks[1];
      int4 v0 = vs[0], v1 = vs[1];
      int4* kd = reinterpret_cast<int4*>(&smK[row * 136 + col]);
      int4* vd = reinterpret_cast<int4*>(&smV[row * 136 + col]);
      kd[0] = k0; kd[1] = k1;
      vd[0] = v0; vd[1] = v1;
      if (jb + 32 < NN) {
        __builtin_prefetch(kbase + (size_t)(jb + 32 + row) * HD + col, 0, 0);
        __builtin_prefetch(vbase + (size_t)(jb + 32 + row) * HD + col, 0, 0);
      }
    }
    __syncthreads();

    // S = Q K^T : two 16x16 tiles over 32 keys. K^T B-operand == A-style read
    // of the [key][d] tile (rows = keys, contiguous along d).
    v8f s0 = {}, s1 = {};
#pragma unroll
    for (int c = 0; c < 4; ++c) {
      v16bf bt0 = ldsA_frag(&smK[0 * 136 + c * 32], 136, lane);
      v16bf bt1 = ldsA_frag(&smK[16 * 136 + c * 32], 136, lane);
      s0 = wmma_bf16(qfrag[c], bt0, s0);
      s1 = wmma_bf16(qfrag[c], bt1, s1);
    }

    // Scale + submatrix mask + online softmax (row = r + 8*half, col = ln).
    const int j0 = jb + ln;
    const int j1 = jb + 16 + ln;
#pragma unroll
    for (int r = 0; r < 8; ++r) {
      float a  = s0[r] * SCALE_;
      float bq = s1[r] * SCALE_;
      const int rowg = qw + r + (half << 3);
      if (rowg < MAXMASK) {
        if (j0 < MAXMASK && mask[rowg * MAXMASK + j0] == 0) a  = -__builtin_inff();
        if (j1 < MAXMASK && mask[rowg * MAXMASK + j1] == 0) bq = -__builtin_inff();
      }
      float vmx = fmaxf(a, bq);
#pragma unroll
      for (int off = 8; off >= 1; off >>= 1) vmx = fmaxf(vmx, __shfl_xor(vmx, off, 32));
      const float mnew  = fmaxf(row_m[r], vmx);
      const float alpha = exp2f((row_m[r] - mnew) * L2E_);
      row_m[r] = mnew;
      const float e0 = exp2f((a - mnew) * L2E_);
      const float e1 = exp2f((bq - mnew) * L2E_);
      float sum = e0 + e1;
#pragma unroll
      for (int off = 8; off >= 1; off >>= 1) sum += __shfl_xor(sum, off, 32);
      row_l[r] = row_l[r] * alpha + sum;
#pragma unroll
      for (int t = 0; t < 8; ++t) oacc[t][r] = oacc[t][r] * alpha;
      __bf16* pw = smP[wave];
      pw[(r + (half << 3)) * 40 + ln]      = f2bf(e0);
      pw[(r + (half << 3)) * 40 + 16 + ln] = f2bf(e1);
    }

    // O += P(16x32) @ V(32x128): per-wave LDS scratch, LDS is in-order per wave.
    v16bf pa = ldsA_frag(smP[wave], 40, lane);
#pragma unroll
    for (int t = 0; t < 8; ++t) {
      v16bf vb = ldsB_frag(&smV[t * 16], 136, lane);
      oacc[t] = wmma_bf16(pa, vb, oacc[t]);
    }
  }

  // Normalize and store O into [b, n, h*d] bf16 (row-major 8192x1024).
  const int b_ = bh >> 3;
  const int h_ = bh & 7;
#pragma unroll
  for (int t = 0; t < 8; ++t)
#pragma unroll
    for (int r = 0; r < 8; ++r) {
      const int rowg = qw + r + (half << 3);
      const int col  = h_ * HD + t * 16 + ln;
      const float val = oacc[t][r] / row_l[r];
      o[((size_t)(b_ * NN + rowg)) * DDIM + col] = f2bf(val);
    }
}

// ---------------------------------------------------------------------------
// Kernel 3: output projection o @ W_out + b_out. Grid (1024/64, 8192/128).
// ---------------------------------------------------------------------------
__global__ __launch_bounds__(256)
void proj_kernel(const __bf16* __restrict__ a, const float* __restrict__ w,
                 const float* __restrict__ bias, float* __restrict__ out) {
  const int tid  = threadIdx.x;
  const int lane = tid & 31;
  const int wave = tid >> 5;
  const int half = lane >> 4;
  const int ln   = lane & 15;
  const int wm   = wave & 3;
  const int wn   = wave >> 2;
  const int rowBase = blockIdx.y * 128;
  const int colBase = blockIdx.x * 64;

  __shared__ __bf16 smA[128 * 40];
  __shared__ __bf16 smB[32 * 72];

  v8f acc[2][2] = {};

  for (int k0 = 0; k0 < DDIM; k0 += 32) {
    __syncthreads();
    { // A tile: 128x32 bf16 copy (16 elems / thread)
      const int row = tid >> 1;
      const int col = (tid & 1) << 4;
      const int4* src =
          reinterpret_cast<const int4*>(a + (size_t)(rowBase + row) * DDIM + k0 + col);
      int4 a0 = src[0], a1 = src[1];
      int4* d = reinterpret_cast<int4*>(&smA[row * 40 + col]);
      d[0] = a0; d[1] = a1;
      if (k0 + 32 < DDIM)
        __builtin_prefetch(a + (size_t)(rowBase + row) * DDIM + k0 + 32 + col, 0, 1);
    }
    { // B tile: 32x64 f32 -> bf16 (8 elems / thread)
      const int row = tid >> 3;
      const int col = (tid & 7) << 3;
      const float4* src =
          reinterpret_cast<const float4*>(w + (size_t)(k0 + row) * DDIM + colBase + col);
#pragma unroll
      for (int q4 = 0; q4 < 2; ++q4) {
        float4 f = src[q4];
        __bf16* d = &smB[row * 72 + col + q4 * 4];
        d[0] = f2bf(f.x); d[1] = f2bf(f.y); d[2] = f2bf(f.z); d[3] = f2bf(f.w);
      }
    }
    __syncthreads();

    v16bf af[2], bfr[2];
#pragma unroll
    for (int i = 0; i < 2; ++i) af[i] = ldsA_frag(&smA[(wm * 32 + i * 16) * 40], 40, lane);
#pragma unroll
    for (int j = 0; j < 2; ++j) bfr[j] = ldsB_frag(&smB[wn * 32 + j * 16], 72, lane);
#pragma unroll
    for (int i = 0; i < 2; ++i)
#pragma unroll
      for (int j = 0; j < 2; ++j) acc[i][j] = wmma_bf16(af[i], bfr[j], acc[i][j]);
  }

  const int hh8 = half << 3;
#pragma unroll
  for (int i = 0; i < 2; ++i)
#pragma unroll
    for (int j = 0; j < 2; ++j) {
      const int c = colBase + wn * 32 + j * 16 + ln;
      const float bv = bias[c];
#pragma unroll
      for (int r = 0; r < 8; ++r) {
        const int m = rowBase + wm * 32 + i * 16 + r + hh8;
        out[(size_t)m * DDIM + c] = acc[i][j][r] + bv;
      }
    }
}

// ---------------------------------------------------------------------------
// Launch. Workspace layout (bf16): q | k | v | o, 16 MB each (64 MB total).
// ---------------------------------------------------------------------------
extern "C" void kernel_launch(void* const* d_in, const int* in_sizes, int n_in,
                              void* d_out, int out_size, void* d_ws, size_t ws_size,
                              hipStream_t stream) {
  const float* x    = (const float*)d_in[0];
  const float* wqkv = (const float*)d_in[1];
  const float* wout = (const float*)d_in[2];
  const float* bout = (const float*)d_in[3];
  const int*   mask = (const int*)d_in[4];

  const size_t tsz = (size_t)BB * HEADS * NN * HD;  // 8,388,608 elements
  __bf16* qws = (__bf16*)d_ws;
  __bf16* kws = qws + tsz;
  __bf16* vws = kws + tsz;
  __bf16* ows = vws + tsz;

  qkv_kernel<<<dim3(K3 / 64, MROWS / 128), 256, 0, stream>>>(x, wqkv, qws, kws, vws);
  attn_kernel<<<dim3(BB * HEADS, NN / 128), 256, 0, stream>>>(qws, kws, vws, mask, ows);
  proj_kernel<<<dim3(DDIM / 64, MROWS / 128), 256, 0, stream>>>(ows, wout, bout,
                                                                (float*)d_out);
}
